// NorseMNISTNet_84782654423175
// MI455X (gfx1250) — matrix operand order repaired and checked by reading, not compiled, not run on hardware
//
#include <hip/hip_runtime.h>
#include <hip/hip_bf16.h>

// ---------------------------------------------------------------------------
// LIF spiking net forward (MI455X / gfx1250):
//   I1 = mean_c(inputs) @ W1   -- fused: A fragments built from raw inputs
//                                 (channel mean + f32->f16) feeding
//                                 v_wmma_f32_16x16x32_f16; B staged in LDS via
//                                 global_load_async_to_lds_b128 (ASYNCcnt).
//   z1 = LIF+BN scan over T    -- per-feature blocks, LDS reductions
//   I2 = z1 @ W2               -- WMMA, K=128pad, N=16pad, LDS-staged B
//   out = mean_t(LIF scan of I2)
// Traffic: ~321 MB input stream dominates (~14 us @ 23.3 TB/s); GEMM math
// (8 GFLOP f16) is negligible on the WMMA pipe, so fusing away the 160 MB
// packed-activation round trip is the main win vs round 1.
// ---------------------------------------------------------------------------

typedef __attribute__((ext_vector_type(16))) _Float16 v16h;
typedef __attribute__((ext_vector_type(8)))  _Float16 v8h;
typedef __attribute__((ext_vector_type(8)))  float    v8f;
typedef __attribute__((ext_vector_type(4)))  float    v4f;
typedef int v4i __attribute__((vector_size(16)));   // matches builtin param type
typedef __attribute__((address_space(1))) v4i* gv4i_p;  // global int4*
typedef __attribute__((address_space(3))) v4i* lv4i_p;  // LDS int4*

#define T_STEPS 100
#define BATCH   512
#define F_IN    784
#define K1PAD   800   // 784 -> 25 * 32
#define N1PAD   112   // 100 -> 7 * 16
#define K2PAD   128   // 100 -> 4 * 32
#define N2PAD   16    // 10  -> 1 * 16
#define ALPHA   0.95f
#define BN_EPS  1e-5f

#if defined(__has_builtin)
#if __has_builtin(__builtin_amdgcn_global_load_async_to_lds_b128)
#define HAVE_ASYNC_LDS 1
#endif
#if __has_builtin(__builtin_amdgcn_s_wait_asynccnt)
#define HAVE_WAIT_ASYNC 1
#endif
#endif

__device__ __forceinline__ void wait_asynccnt0()
{
#if defined(HAVE_WAIT_ASYNC)
    __builtin_amdgcn_s_wait_asynccnt(0);
#else
    asm volatile("s_wait_asynccnt 0x0" ::: "memory");
#endif
}

// Stage a 32-wide K slice of Bt[NT*16][K] into LDS (layout [n][32]).
// 16-byte chunks: NT*16 rows x 4 chunks. All 256 threads cooperate.
template<int NT>
__device__ __forceinline__ void stage_B(const _Float16* __restrict__ Bt, int K,
                                        int k0, _Float16* dstLds)
{
    constexpr int CHUNKS = NT * 16 * 4;
    for (int c = threadIdx.x; c < CHUNKS; c += 256) {
        const int n = c >> 2, part = c & 3;
        const _Float16* src = Bt + (size_t)n * K + k0 + part * 8;
        _Float16* dst = dstLds + n * 32 + part * 8;
#if defined(HAVE_ASYNC_LDS)
        __builtin_amdgcn_global_load_async_to_lds_b128(
            (gv4i_p)(v4i*)src, (lv4i_p)(v4i*)dst, 0, 0);
#else
        *(v8h*)dst = *(const v8h*)src;
#endif
    }
}

// Build 8 f16 A values = channel means of 8 consecutive features starting at f
// (16 consecutive floats, 64 B, 16 B aligned). f is 8-aligned so the chunk is
// uniformly valid iff f < 784; otherwise zero (K padding).
__device__ __forceinline__ v8h load_mean8(const float* __restrict__ inrow, int f)
{
    v8h h{};
    if (f < F_IN) {
        const v4f* p = (const v4f*)(inrow + (size_t)f * 2);
        v4f c0 = p[0], c1 = p[1], c2 = p[2], c3 = p[3];
        h[0] = (_Float16)(0.5f * (c0[0] + c0[1]));
        h[1] = (_Float16)(0.5f * (c0[2] + c0[3]));
        h[2] = (_Float16)(0.5f * (c1[0] + c1[1]));
        h[3] = (_Float16)(0.5f * (c1[2] + c1[3]));
        h[4] = (_Float16)(0.5f * (c2[0] + c2[1]));
        h[5] = (_Float16)(0.5f * (c2[2] + c2[3]));
        h[6] = (_Float16)(0.5f * (c3[0] + c3[1]));
        h[7] = (_Float16)(0.5f * (c3[2] + c3[3]));
    }
    return h;
}

// ---------------------------------------------------------------------------
// GEMM1 fused: C[M,112] = mean_c(inputs) @ W1t^T, A built on the fly.
// One wave per 16-row strip (rows = t*B+b; B%16==0 so t is wave-uniform).
// A fragment layout (16-bit A 16x32, wave32):
//   lanes 0-15 : row=lane,    K = k0+[0..7]  and k0+[16..23]
//   lanes16-31 : row=lane-16, K = k0+[8..15] and k0+[24..31]
// B fragment (16-bit B 32x16): lane holds col=lane%16, same K as A:
//   k0 + (lane/16)*8 + [0..7] and +16 -> two ds_load_b128 from the staged slice.
// ---------------------------------------------------------------------------
__global__ __launch_bounds__(256)
void gemm1_fused_kernel(const float* __restrict__ in,     // [B,T,784,2]
                        const _Float16* __restrict__ Bt,  // [112,800]
                        float* __restrict__ C)            // [M,112]
{
    constexpr int NT = 7;
    constexpr int K  = K1PAD;
    const int wave = threadIdx.x >> 5;
    const int lane = threadIdx.x & 31;
    const int half = lane >> 4;
    const int r    = lane & 15;
    const int m0   = (blockIdx.x * 8 + wave) * 16;
    const int m    = m0 + r;
    const int t    = m / BATCH;        // wave-uniform
    const int b    = m % BATCH;
    const float* inrow = in + ((size_t)b * T_STEPS + t) * (F_IN * 2);

    __shared__ alignas(16) _Float16 Bs[2][NT * 16 * 32];

    stage_B<NT>(Bt, K, 0, &Bs[0][0]);
    wait_asynccnt0();
    __syncthreads();

    v8f zero{};
    v8f acc[NT];
#pragma unroll
    for (int i = 0; i < NT; ++i) acc[i] = zero;

    int buf = 0;
    for (int k0 = 0; k0 < K; k0 += 32) {
        if (k0 + 32 < K) stage_B<NT>(Bt, K, k0 + 32, &Bs[buf ^ 1][0]);
        __builtin_prefetch(inrow + (size_t)(k0 + 32) * 2, 0, 0);

        const int fA = k0 + half * 8;
        v8h a0 = load_mean8(inrow, fA);
        v8h a1 = load_mean8(inrow, fA + 16);
        v16h a;
#pragma unroll
        for (int i = 0; i < 8; ++i) { a[i] = a0[i]; a[i + 8] = a1[i]; }

#pragma unroll
        for (int nt = 0; nt < NT; ++nt) {
            const _Float16* pb = &Bs[buf][(nt * 16 + r) * 32 + half * 8];
            v8h b0 = *(const v8h*)(pb);
            v8h b1 = *(const v8h*)(pb + 16);
            v16h bb;
#pragma unroll
            for (int i = 0; i < 8; ++i) { bb[i] = b0[i]; bb[i + 8] = b1[i]; }
            acc[nt] = __builtin_amdgcn_wmma_f32_16x16x32_f16(
                false, a, false, bb, (short)0, acc[nt], false, false);
        }

        wait_asynccnt0();   // staged slice for next iter landed
        __syncthreads();    // all waves done reading buf; flip
        buf ^= 1;
    }

    // C layout: VGPR i -> row m0 + half*8 + i, col = n0 + r
#pragma unroll
    for (int nt = 0; nt < NT; ++nt) {
#pragma unroll
        for (int i = 0; i < 8; ++i) {
            int row = m0 + half * 8 + i;
            C[(size_t)row * N1PAD + nt * 16 + r] = acc[nt][i];
        }
    }
}

// ---------------------------------------------------------------------------
// Generic WMMA GEMM with LDS-staged B: C[M,NT*16] = A[M,K] @ Bt[NT*16,K]^T
// (used for GEMM2: A = spikes f16, K=128, NT=1)
// ---------------------------------------------------------------------------
template<int NT>
__global__ __launch_bounds__(256)
void gemm_wmma_lds_kernel(const _Float16* __restrict__ A,
                          const _Float16* __restrict__ Bt,
                          float* __restrict__ C,
                          int M, int K)
{
    const int wave = threadIdx.x >> 5;
    const int lane = threadIdx.x & 31;
    const int half = lane >> 4;
    const int r    = lane & 15;
    const int m0   = (blockIdx.x * 8 + wave) * 16;
    const _Float16* arow = A + (size_t)(m0 + r) * K;

    __shared__ alignas(16) _Float16 Bs[2][NT * 16 * 32];

    stage_B<NT>(Bt, K, 0, &Bs[0][0]);
    wait_asynccnt0();
    __syncthreads();

    v8f zero{};
    v8f acc[NT];
#pragma unroll
    for (int i = 0; i < NT; ++i) acc[i] = zero;

    int buf = 0;
    for (int k0 = 0; k0 < K; k0 += 32) {
        if (k0 + 32 < K) stage_B<NT>(Bt, K, k0 + 32, &Bs[buf ^ 1][0]);

        const _Float16* pa = arow + k0 + half * 8;
        v8h a0 = *(const v8h*)(pa);
        v8h a1 = *(const v8h*)(pa + 16);
        v16h a;
#pragma unroll
        for (int i = 0; i < 8; ++i) { a[i] = a0[i]; a[i + 8] = a1[i]; }

#pragma unroll
        for (int nt = 0; nt < NT; ++nt) {
            const _Float16* pb = &Bs[buf][(nt * 16 + r) * 32 + half * 8];
            v8h b0 = *(const v8h*)(pb);
            v8h b1 = *(const v8h*)(pb + 16);
            v16h bb;
#pragma unroll
            for (int i = 0; i < 8; ++i) { bb[i] = b0[i]; bb[i + 8] = b1[i]; }
            acc[nt] = __builtin_amdgcn_wmma_f32_16x16x32_f16(
                false, a, false, bb, (short)0, acc[nt], false, false);
        }

        wait_asynccnt0();
        __syncthreads();
        buf ^= 1;
    }

    const int N = NT * 16;
#pragma unroll
    for (int nt = 0; nt < NT; ++nt) {
#pragma unroll
        for (int i = 0; i < 8; ++i) {
            int row = m0 + half * 8 + i;
            C[(size_t)row * N + nt * 16 + r] = acc[nt][i];
        }
    }
}

// ---------------------------------------------------------------------------
// Weight packing (transpose + f16 + zero pad) and spike-buffer zero fill
// ---------------------------------------------------------------------------
__global__ __launch_bounds__(256)
void pack_w1t_kernel(const float* __restrict__ W1, _Float16* __restrict__ W1t)
{
    int idx = blockIdx.x * blockDim.x + threadIdx.x;   // over N1PAD*K1PAD
    if (idx >= N1PAD * K1PAD) return;
    int n = idx / K1PAD;
    int k = idx % K1PAD;
    float v = (n < 100 && k < F_IN) ? W1[k * 100 + n] : 0.0f;
    W1t[idx] = (_Float16)v;
}

__global__ __launch_bounds__(256)
void pack_w2t_kernel(const float* __restrict__ W2, _Float16* __restrict__ W2t)
{
    int idx = blockIdx.x * blockDim.x + threadIdx.x;   // over N2PAD*K2PAD
    if (idx >= N2PAD * K2PAD) return;
    int n = idx / K2PAD;
    int k = idx % K2PAD;
    float v = (n < 10 && k < 100) ? W2[k * 10 + n] : 0.0f;
    W2t[idx] = (_Float16)v;
}

__global__ __launch_bounds__(256)
void fill_zero_h_kernel(_Float16* __restrict__ p, long long n)
{
    long long idx = (long long)blockIdx.x * blockDim.x + threadIdx.x;
    if (idx < n) p[idx] = (_Float16)0.0f;
}

// ---------------------------------------------------------------------------
// LIF layer 1 with per-timestep batch norm (batch stats over B=512).
// One block per output feature o; one thread per batch element.
// ---------------------------------------------------------------------------
__global__ __launch_bounds__(512)
void lif1_kernel(const float* __restrict__ I1,
                 const float* __restrict__ scale,
                 const float* __restrict__ bias,
                 _Float16* __restrict__ z1h)
{
    const int o = blockIdx.x;      // 0..99
    const int b = threadIdx.x;     // 0..511
    __shared__ float red[512];

    const float sc = scale[o];
    const float bi = bias[o];
    float v = 0.0f;

    for (int t = 0; t < T_STEPS; ++t) {
        float h = I1[((size_t)t * BATCH + b) * N1PAD + o];

        red[b] = h;
        __syncthreads();
        for (int s = 256; s > 0; s >>= 1) {
            if (b < s) red[b] += red[b + s];
            __syncthreads();
        }
        float mu = red[0] * (1.0f / BATCH);
        __syncthreads();

        float d = h - mu;
        red[b] = d * d;
        __syncthreads();
        for (int s = 256; s > 0; s >>= 1) {
            if (b < s) red[b] += red[b + s];
            __syncthreads();
        }
        float var = red[0] * (1.0f / BATCH);
        __syncthreads();

        float hn = d * rsqrtf(var + BN_EPS) * sc + bi;

        v = ALPHA * v + hn;
        float z = (v > 1.0f) ? 1.0f : 0.0f;
        v *= (1.0f - z);

        z1h[((size_t)t * BATCH + b) * K2PAD + o] = (_Float16)z;
    }
}

// ---------------------------------------------------------------------------
// LIF layer 2 (no BN) + mean over time. One thread per (b, class).
// ---------------------------------------------------------------------------
__global__ __launch_bounds__(256)
void lif2_kernel(const float* __restrict__ I2, float* __restrict__ out)
{
    int idx = blockIdx.x * blockDim.x + threadIdx.x;   // b*10 + j
    if (idx >= BATCH * 10) return;
    int b = idx / 10;
    int j = idx % 10;

    float v = 0.0f, s = 0.0f;
    for (int t = 0; t < T_STEPS; ++t) {
        float i = I2[((size_t)t * BATCH + b) * N2PAD + j];
        v = ALPHA * v + i;
        float z = (v > 1.0f) ? 1.0f : 0.0f;
        v *= (1.0f - z);
        s += z;
    }
    out[idx] = s * (1.0f / T_STEPS);
}

// ---------------------------------------------------------------------------
// Launch
// ---------------------------------------------------------------------------
extern "C" void kernel_launch(void* const* d_in, const int* in_sizes, int n_in,
                              void* d_out, int out_size, void* d_ws, size_t ws_size,
                              hipStream_t stream)
{
    const float* inputs   = (const float*)d_in[0];   // [512,100,28,28,2]
    // d_in[1] = trgt (unused in forward)
    const float* W1       = (const float*)d_in[2];   // [784,100]
    const float* bn_scale = (const float*)d_in[3];   // [100]
    const float* bn_bias  = (const float*)d_in[4];   // [100]
    const float* W2       = (const float*)d_in[5];   // [100,10]
    float* out = (float*)d_out;                      // [512,10]

    const long long M = (long long)T_STEPS * BATCH;  // 51200

    // workspace carve-up (all offsets multiple of 256 bytes)
    char* ws = (char*)d_ws;
    size_t off = 0;
    _Float16* W1t = (_Float16*)(ws + off); off += (size_t)N1PAD * K1PAD * 2;    // 179,200
    _Float16* W2t = (_Float16*)(ws + off); off += (size_t)N2PAD * K2PAD * 2;    // 4,096
    float*    I1  = (float*)   (ws + off); off += (size_t)M * N1PAD * 4;        // 22,937,600
    _Float16* z1h = (_Float16*)(ws + off); off += (size_t)M * K2PAD * 2;        // 13,107,200
    float*    I2  = (float*)   (ws + off); off += (size_t)M * N2PAD * 4;        // 3,276,800
    (void)ws_size;

    // 1) pack transposed, padded weights
    pack_w1t_kernel<<<(N1PAD * K1PAD + 255) / 256, 256, 0, stream>>>(W1, W1t);
    pack_w2t_kernel<<<(N2PAD * K2PAD + 255) / 256, 256, 0, stream>>>(W2, W2t);
    // 2) zero spike buffer (pad columns 100..127 must be 0 for GEMM2)
    {
        long long n = M * K2PAD;
        fill_zero_h_kernel<<<(int)((n + 255) / 256), 256, 0, stream>>>(z1h, n);
    }
    // 3) GEMM1 fused with channel-mean + f16 convert: I1 = x @ W1
    gemm1_fused_kernel<<<(int)(M / 128), 256, 0, stream>>>(inputs, W1t, I1);
    // 4) LIF layer 1 with batch norm -> spikes (f16)
    lif1_kernel<<<100, 512, 0, stream>>>(I1, bn_scale, bn_bias, z1h);
    // 5) GEMM2: I2[51200,16] = z1h[51200,128] @ W2t^T
    gemm_wmma_lds_kernel<1><<<(int)(M / 128), 256, 0, stream>>>(z1h, W2t, I2, (int)M, K2PAD);
    // 6) LIF layer 2 + time mean
    lif2_kernel<<<(BATCH * 10 + 255) / 256, 256, 0, stream>>>(I2, out);
    (void)out_size; (void)in_sizes; (void)n_in;
}
